// SepSpacialAttention_51651276702313
// MI455X (gfx1250) — matrix operand order repaired and checked by reading, not compiled, not run on hardware
//
#include <hip/hip_runtime.h>
#include <hip/hip_bf16.h>

// Problem constants
#define B_     32
#define S_     128
#define J_     24
#define D_     256
#define H_     4
#define DEPTH_ 64
#define N_     (B_ * S_)            // 4096 (b,s) pairs

typedef __attribute__((ext_vector_type(16))) __bf16 v16bf;
typedef __attribute__((ext_vector_type(8)))  float  v8f;

union BfPack { unsigned int u[8]; v16bf v; };

// ---- scratch layout (in ushort elements) ----
//   Wk @ 0        (65536)      bf16 weights
//   Wv @ 65536    (65536)
//   Wo @ 131072   (65536)
//   Wq @ 196608   (1572864)
//   Q  @ 1769472  (98304*256)  bf16 activations
//   K  @ 26935296
//   V  @ 52101120
// total = 77266944 ushorts = ~155 MB
#define WS_WK 0
#define WS_WV 65536
#define WS_WO 131072
#define WS_WQ 196608
#define WS_Q  1769472
#define WS_K  26935296
#define WS_V  52101120

// Native f32 -> bf16 convert (hardware v_cvt on gfx1250).
__device__ __forceinline__ unsigned short f2bf(float f) {
  union { __bf16 h; unsigned short u; } c;
  c.h = (__bf16)f;
  return c.u;
}

// Packed pair convert: two adjacent f32 -> one dword of two bf16
// (compiler can fuse into v_cvt_pk_bf16_f32).
__device__ __forceinline__ unsigned int f2bf2(float lo, float hi) {
  return (unsigned int)f2bf(lo) | ((unsigned int)f2bf(hi) << 16);
}

// A operand (16-bit, 16x32): lane = M row; lanes 0-15 hold K 0..15,
// lanes 16-31 hold K 16..31 (pairs per VGPR). Matrix row-major, ld even.
__device__ __forceinline__ v16bf loadA(const unsigned short* M, int ld, int m0, int k0, int lane) {
  BfPack p;
  const unsigned short* row = M + (m0 + (lane & 15)) * ld + k0;
  const int kh = (lane >> 4) << 4;
#pragma unroll
  for (int v = 0; v < 8; ++v) {
    int k = 2 * v + kh;
    p.u[v] = *(const unsigned int*)(row + k);
  }
  return p.v;
}

// B operand (16-bit, 32x16) from a matrix stored [N][K] row-major (K-pairs
// contiguous): lane = N col; lanes 0-15 hold K 0..15, lanes 16-31 K 16..31.
__device__ __forceinline__ v16bf loadB_wt(const unsigned short* M, int ld, int n0, int k0, int lane) {
  BfPack p;
  const unsigned short* row = M + (n0 + (lane & 15)) * ld + k0 + ((lane >> 4) << 4);
#pragma unroll
  for (int v = 0; v < 8; ++v) p.u[v] = *(const unsigned int*)(row + 2 * v);
  return p.v;
}

// B operand from a matrix stored [K][N] row-major (contraction along rows):
// two 16-bit gathers per VGPR.
__device__ __forceinline__ v16bf loadB_kmaj(const unsigned short* M, int ld, int n0, int lane) {
  BfPack p;
  const int n  = n0 + (lane & 15);
  const int kh = (lane >> 4) << 4;
#pragma unroll
  for (int v = 0; v < 8; ++v) {
    unsigned int lo = M[(kh + 2 * v) * ld + n];
    unsigned int hi = M[(kh + 2 * v + 1) * ld + n];
    p.u[v] = lo | (hi << 16);
  }
  return p.v;
}

__device__ __forceinline__ v8f wmma_bf16(v16bf a, v16bf b, v8f c) {
  return __builtin_amdgcn_wmma_f32_16x16x32_bf16(false, a, false, b, (short)0, c, false, false);
}

// ---------------- Kernel 0: weights f32 -> bf16 (packed pairs) ----------------
__global__ void convert_weights_kernel(const float* __restrict__ Wk, const float* __restrict__ Wv,
                                       const float* __restrict__ Wq, const float* __restrict__ Wo,
                                       unsigned short* __restrict__ ws) {
  unsigned int* wsw = (unsigned int*)ws;   // all region offsets are even
  const int totalw = (3 * 65536 + 24 * 65536) / 2;   // dword pairs
  for (int i = blockIdx.x * blockDim.x + threadIdx.x; i < totalw; i += gridDim.x * blockDim.x) {
    const int e = 2 * i;
    const float2* src;
    unsigned int* dst;
    if      (e <  65536) { src = (const float2*)(Wk + e);           dst = wsw + (WS_WK + e) / 2; }
    else if (e < 131072) { src = (const float2*)(Wv + e -  65536);  dst = wsw + (WS_WV + e - 65536) / 2; }
    else if (e < 196608) { src = (const float2*)(Wo + e - 131072);  dst = wsw + (WS_WO + e - 131072) / 2; }
    else                 { src = (const float2*)(Wq + e - 196608);  dst = wsw + (WS_WQ + e - 196608) / 2; }
    float2 v = *src;
    *dst = f2bf2(v.x, v.y);
  }
}

// ---------------- Kernel 1: Q/K/V projections ----------------
// grid = (N_/64) * 24 blocks; block = fixed j, 64 consecutive n rows.
// Computes q = x@Wq[j]^T + bq[j], k = x@Wk^T + bk, v = x@Wv^T + bv (bf16 out).
__global__ void qkv_kernel(const float* __restrict__ x,
                           const float* __restrict__ bk, const float* __restrict__ bvp,
                           const float* __restrict__ bq,
                           const unsigned short* __restrict__ ws,
                           unsigned short* __restrict__ wsQ,
                           unsigned short* __restrict__ wsK,
                           unsigned short* __restrict__ wsV) {
  extern __shared__ unsigned short Xb[];   // [64][258] bf16 (padded vs bank conflicts)
  const int LD   = 258;
  const int tid  = threadIdx.x;
  const int lane = tid & 31, wave = tid >> 5;
  const int j  = blockIdx.x % J_;
  const int n0 = (blockIdx.x / J_) * 64;

  // Stage x tile -> LDS as bf16 (2 elements / iteration, packed converts)
  for (int i = tid; i < 64 * (D_ / 2); i += 256) {
    const int r = i >> 7, c2 = i & 127;
    float2 v = *(const float2*)(x + ((size_t)(n0 + r) * J_ + j) * D_ + 2 * c2);
    *(unsigned int*)(Xb + r * LD + 2 * c2) = f2bf2(v.x, v.y);
  }
  __syncthreads();

  const unsigned short* Wq = ws + WS_WQ + j * (D_ * D_);
  const unsigned short* Wk = ws + WS_WK;
  const unsigned short* Wv = ws + WS_WV;

  // 3 mats * 4 Mtiles * 16 Ntiles = 192 tiles; 24 per wave (wave-uniform).
#pragma unroll 1
  for (int i = 0; i < 24; ++i) {
    const int t = wave * 24 + i;
    const int mat = t >> 6, rem = t & 63;
    const int mtile = rem >> 4, ntile = rem & 15;
    const unsigned short* W    = (mat == 0) ? Wq : (mat == 1) ? Wk : Wv;
    const float*          bias = (mat == 0) ? (bq + j * D_) : (mat == 1) ? bk : bvp;
    unsigned short*       dst  = (mat == 0) ? wsQ : (mat == 1) ? wsK : wsV;

    const float bias_e = bias[ntile * 16 + (lane & 15)];
    v8f acc;
#pragma unroll
    for (int r = 0; r < 8; ++r) acc[r] = bias_e;

#pragma unroll
    for (int kk = 0; kk < 8; ++kk) {
      v16bf a = loadA(Xb, LD, mtile * 16, kk * 32, lane);
      v16bf b = loadB_wt(W, D_, ntile * 16, kk * 32, lane);
      acc = wmma_bf16(a, b, acc);
    }
    const int e     = ntile * 16 + (lane & 15);
    const int mbase = mtile * 16 + ((lane >> 4) << 3);
#pragma unroll
    for (int r = 0; r < 8; ++r) {
      const int nrow = n0 + mbase + r;
      dst[((size_t)nrow * J_ + j) * D_ + e] = f2bf(acc[r]);
    }
  }
}

// ---------------- Kernel 2: attention + output projection ----------------
// One block per n = b*S + s. 256 threads = 8 waves.
__global__ void attn_kernel(const float* __restrict__ bo,
                            const unsigned short* __restrict__ ws,
                            const unsigned short* __restrict__ wsQ,
                            const unsigned short* __restrict__ wsK,
                            const unsigned short* __restrict__ wsV,
                            float* __restrict__ out, float* __restrict__ attnOut) {
  extern __shared__ unsigned char smem[];
  const int LD = 258;
  unsigned short* Qb = (unsigned short*)smem;       // [32][258] bf16
  unsigned short* Kb = Qb + 32 * LD;
  unsigned short* Vb = Kb + 32 * LD;
  float*          lg = (float*)(smem + (size_t)3 * 32 * LD * 2);       // [4][32][32] f32
  unsigned short* at = (unsigned short*)(lg + 4 * 32 * 32);            // [4][32][32] bf16
  unsigned short* Ob = Qb;  // Qb dead after logits stage -> reuse for attn output

  const int tid = threadIdx.x, lane = tid & 31, wave = tid >> 5;
  const int n = blockIdx.x;

  // Load Q/K/V rows 0..23 (as dwords), zero-pad rows 24..31.
  for (int i = tid; i < 3 * 24 * 128; i += 256) {
    const int arr = i / (24 * 128), rem = i % (24 * 128);
    const int r = rem / 128, c = rem % 128;
    const unsigned int* src = (arr == 0) ? (const unsigned int*)wsQ
                            : (arr == 1) ? (const unsigned int*)wsK
                                         : (const unsigned int*)wsV;
    unsigned short* dst = (arr == 0) ? Qb : (arr == 1) ? Kb : Vb;
    *(unsigned int*)(dst + r * LD + 2 * c) = src[((size_t)n * J_ + r) * 128 + c];
  }
  for (int i = tid; i < 3 * 8 * 129; i += 256) {
    const int arr = i / (8 * 129), rem = i % (8 * 129);
    const int r = 24 + rem / 129, c = rem % 129;
    unsigned short* dst = (arr == 0) ? Qb : (arr == 1) ? Kb : Vb;
    *(unsigned int*)(dst + r * LD + 2 * c) = 0u;
  }
  __syncthreads();

  // logits[h][j][k] = (Q_h K_h^T)/8 : 4 heads * 2x2 tiles = 16 jobs, 2/wave.
#pragma unroll 1
  for (int i = 0; i < 2; ++i) {
    const int t = wave * 2 + i;
    const int h = t >> 2, mtile = (t >> 1) & 1, ntile = t & 1;
    v8f acc;
#pragma unroll
    for (int r = 0; r < 8; ++r) acc[r] = 0.f;
#pragma unroll
    for (int kk = 0; kk < 2; ++kk) {
      v16bf a = loadA(Qb, LD, mtile * 16, h * DEPTH_ + kk * 32, lane);
      v16bf b = loadB_wt(Kb, LD, ntile * 16, h * DEPTH_ + kk * 32, lane);
      acc = wmma_bf16(a, b, acc);
    }
    float* lbase = lg + h * 1024 + ntile * 16 + (lane & 15);
    const int mbase = mtile * 16 + ((lane >> 4) << 3);
#pragma unroll
    for (int r = 0; r < 8; ++r) lbase[(mbase + r) * 32] = acc[r] * 0.125f;
  }
  __syncthreads();

  // Softmax over the 24 real keys; pad columns zeroed (keeps NaN out of AV).
  if (tid < 128) {
    const int h = tid >> 5, jr = tid & 31;
    float* row = lg + (h * 32 + jr) * 32;
    float m = row[0];
    for (int k = 1; k < 24; ++k) m = fmaxf(m, row[k]);
    float p[24], ssum = 0.f;
    for (int k = 0; k < 24; ++k) { p[k] = __expf(row[k] - m); ssum += p[k]; }
    const float inv = 1.f / ssum;
    unsigned short* arow = at + (h * 32 + jr) * 32;
    for (int k = 0; k < 24; k += 2) {
      float p0 = p[k] * inv, p1 = p[k + 1] * inv;
      *(unsigned int*)(arow + k) = f2bf2(p0, p1);
      row[k] = p0; row[k + 1] = p1;
    }
    *(unsigned int*)(arow + 24) = 0u;
    *(unsigned int*)(arow + 26) = 0u;
    *(unsigned int*)(arow + 28) = 0u;
    *(unsigned int*)(arow + 30) = 0u;
    // attention_weights output for s == S-1 (f32)
    if ((n % S_) == (S_ - 1) && jr < J_) {
      const int bb = n / S_;
      float* dsta = attnOut + (((size_t)bb * H_ + h) * J_ + jr) * J_;
      for (int k = 0; k < 24; ++k) dsta[k] = row[k];
    }
  }
  __syncthreads();

  // out_h = attn_h @ V_h : 32 tiles (4h * 2m * 4n), 4/wave, K=32 (padded).
#pragma unroll 1
  for (int i = 0; i < 4; ++i) {
    const int t = wave * 4 + i;
    const int h = t >> 3, rem = t & 7, mtile = rem >> 2, nsub = rem & 3;
    v8f acc;
#pragma unroll
    for (int r = 0; r < 8; ++r) acc[r] = 0.f;
    v16bf a = loadA(at + h * 1024, 32, mtile * 16, 0, lane);
    v16bf b = loadB_kmaj(Vb, LD, h * DEPTH_ + nsub * 16, lane);
    acc = wmma_bf16(a, b, acc);
    const int col   = h * DEPTH_ + nsub * 16 + (lane & 15);
    const int mbase = mtile * 16 + ((lane >> 4) << 3);
#pragma unroll
    for (int r = 0; r < 8; ++r) Ob[(mbase + r) * LD + col] = f2bf(acc[r]);
  }
  __syncthreads();

  // output = Ob @ Wo^T + bo : 32 tiles, 4/wave; store only rows < 24.
  const unsigned short* Wo = ws + WS_WO;
#pragma unroll 1
  for (int i = 0; i < 4; ++i) {
    const int t = wave * 4 + i;
    const int mtile = t >> 4, ntile = t & 15;
    const float bias_e = bo[ntile * 16 + (lane & 15)];
    v8f acc;
#pragma unroll
    for (int r = 0; r < 8; ++r) acc[r] = bias_e;
#pragma unroll
    for (int kk = 0; kk < 8; ++kk) {
      v16bf a = loadA(Ob, LD, mtile * 16, kk * 32, lane);
      v16bf b = loadB_wt(Wo, D_, ntile * 16, kk * 32, lane);
      acc = wmma_bf16(a, b, acc);
    }
    const int e     = ntile * 16 + (lane & 15);
    const int mbase = mtile * 16 + ((lane >> 4) << 3);
#pragma unroll
    for (int r = 0; r < 8; ++r) {
      const int jrow = mbase + r;
      if (jrow < J_) out[((size_t)n * J_ + jrow) * D_ + e] = acc[r];
    }
  }
}

extern "C" void kernel_launch(void* const* d_in, const int* in_sizes, int n_in,
                              void* d_out, int out_size, void* d_ws, size_t ws_size,
                              hipStream_t stream) {
  (void)in_sizes; (void)n_in; (void)out_size; (void)ws_size;
  const float* x  = (const float*)d_in[0];
  const float* Wk = (const float*)d_in[1];
  const float* bk = (const float*)d_in[2];
  const float* Wv = (const float*)d_in[3];
  const float* bv = (const float*)d_in[4];
  const float* Wq = (const float*)d_in[5];
  const float* bq = (const float*)d_in[6];
  const float* Wo = (const float*)d_in[7];
  const float* bo = (const float*)d_in[8];

  unsigned short* ws  = (unsigned short*)d_ws;
  unsigned short* wsQ = ws + WS_Q;
  unsigned short* wsK = ws + WS_K;
  unsigned short* wsV = ws + WS_V;

  float* out     = (float*)d_out;
  float* attnOut = out + (size_t)B_ * S_ * J_ * D_;

  hipLaunchKernelGGL(convert_weights_kernel, dim3(512), dim3(256), 0, stream,
                     Wk, Wv, Wq, Wo, ws);

  hipLaunchKernelGGL(qkv_kernel, dim3((N_ / 64) * J_), dim3(256),
                     64 * 258 * sizeof(unsigned short), stream,
                     x, bk, bv, bq, ws, wsQ, wsK, wsV);

  const size_t smem2 = (size_t)3 * 32 * 258 * 2 + 4 * 32 * 32 * 4 + 4 * 32 * 32 * 2;
  hipLaunchKernelGGL(attn_kernel, dim3(N_), dim3(256), smem2, stream,
                     bo, ws, wsQ, wsK, wsV, out, attnOut);
}
